// Rnn_27230092656897
// MI455X (gfx1250) — matrix operand order repaired and checked by reading, not compiled, not run on hardware
//
#include <hip/hip_runtime.h>

#define H_    128
#define G4_   512
#define IN_   12
#define T_    2048
#define BM_   16
#define HPAD_ 136   // 128 + 8 pad -> row stride 272B, bank-conflict-free frag reads
#define OUT_  4

typedef __attribute__((ext_vector_type(16))) __bf16 v16bf;
typedef __attribute__((ext_vector_type(8)))  __bf16 v8bf;
typedef __attribute__((ext_vector_type(8)))  float  v8f;

__device__ __forceinline__ v8f wmma_bf16(v16bf a, v16bf b, v8f c) {
  return __builtin_amdgcn_wmma_f32_16x16x32_bf16(false, a, false, b, (short)0, c,
                                                 false, false);
}

__device__ __forceinline__ float rcp_(float x) { return __builtin_amdgcn_rcpf(x); }
__device__ __forceinline__ float sig_(float x) { return rcp_(1.0f + __expf(-x)); }
__device__ __forceinline__ float th_(float x) {
  float e = __expf(2.0f * x);
  return (e - 1.0f) * rcp_(e + 1.0f);
}

__device__ __forceinline__ v16bf cat16(v8bf lo, v8bf hi) {
  v16bf r;
#pragma unroll
  for (int i = 0; i < 8; ++i) { r[i] = lo[i]; r[8 + i] = hi[i]; }
  return r;
}

// A-matrix 16x32 bf16 fragment: lanes 0-15 row=lane K={0..15},
// lanes 16-31 row=lane-16 K={16..31}; rowp = this lane's row in LDS.
__device__ __forceinline__ v16bf ldA(const __bf16* rowp, int kt, int hiH) {
  int klo = kt * 32 + (hiH ? 16 : 0);
  int khi = klo + 8;
  return cat16(*(const v8bf*)(rowp + klo), *(const v8bf*)(rowp + khi));
}

// B-matrix 32x16 bf16 fragment from row-major-packed LDS W[n][k] (gate = h@W^T):
// lane n holds column n, K = k0..k0+15 (k0 already includes +16 for hi lanes).
__device__ __forceinline__ v16bf ldB(const __bf16* W, int n, int k0) {
  const __bf16* p = W + n * HPAD_ + k0;
  return cat16(*(const v8bf*)p, *(const v8bf*)(p + 8));
}

extern "C" __global__ __launch_bounds__(256, 1)
void lstm2_fused(const float* __restrict__ xp,
                 const float* __restrict__ Wih0, const float* __restrict__ Whh0,
                 const float* __restrict__ bih0, const float* __restrict__ bhh0,
                 const float* __restrict__ Wih1, const float* __restrict__ Whh1,
                 const float* __restrict__ bih1, const float* __restrict__ bhh1,
                 const float* __restrict__ Wout, const float* __restrict__ bout,
                 float* __restrict__ out)
{
  extern __shared__ __align__(16) unsigned char smem[];
  __bf16* W0   = (__bf16*)smem;            // [512][136] bf16
  __bf16* W1   = W0 + G4_ * HPAD_;         // [512][136]
  __bf16* xbuf = W1 + G4_ * HPAD_;         // [16][32] zero-padded K
  __bf16* h0b  = xbuf + BM_ * 32;          // [16][136]
  __bf16* h1b  = h0b + BM_ * HPAD_;        // [16][136]

  const int tid  = threadIdx.x;
  const int wv   = tid >> 5;
  const int ln   = tid & 31;
  const int lm   = ln & 15;
  const int hiH  = (ln >> 4) & 1;
  const int wcol = wv * 16;                // this wave's 16 h-columns
  const int b0   = blockIdx.x * BM_;       // batch tile base

  // ---- pack recurrent weights into LDS as bf16 (padded rows) ----
  for (int idx = tid; idx < G4_ * H_; idx += 256) {
    int r = idx >> 7, c = idx & 127;
    W0[r * HPAD_ + c] = (__bf16)Whh0[idx];
    W1[r * HPAD_ + c] = (__bf16)Whh1[idx];
  }
  for (int idx = tid; idx < BM_ * HPAD_; idx += 256) {
    h0b[idx] = (__bf16)0.f;
    h1b[idx] = (__bf16)0.f;
  }
  for (int idx = tid; idx < BM_ * 32; idx += 256) xbuf[idx] = (__bf16)0.f;

  // ---- per-wave register-resident input-projection fragments ----
  const int kof = hiH ? 16 : 0;
  v16bf fih0[4];       // W_ih_l0 [512,12], K zero-padded to 32
  v16bf fih1[4][4];    // W_ih_l1 [512,128]
  float bias0v[4], bias1v[4];
#pragma unroll
  for (int g = 0; g < 4; ++g) {
    const int n = g * H_ + wcol + lm;
#pragma unroll
    for (int kk = 0; kk < 16; ++kk) {
      int k = kof + kk;
      fih0[g][kk] = (k < IN_) ? (__bf16)Wih0[n * IN_ + k] : (__bf16)0.f;
    }
#pragma unroll
    for (int kt = 0; kt < 4; ++kt)
#pragma unroll
      for (int kk = 0; kk < 16; ++kk)
        fih1[g][kt][kk] = (__bf16)Wih1[n * H_ + kt * 32 + kof + kk];
    bias0v[g] = bih0[n] + bhh0[n];
    bias1v[g] = bih1[n] + bhh1[n];
  }

  v8f c0 = {}, c1 = {};
  const int hcol = wcol + lm;     // D-layout: this lane's column
  const int rb   = hiH ? 8 : 0;   // D-layout: row base for this half-wave
  const __bf16* h0row = h0b + lm * HPAD_;
  const __bf16* h1row = h1b + lm * HPAD_;
  const __bf16* xrow  = xbuf + lm * 32;

  __syncthreads();

  for (int t = 0; t < T_; ++t) {
    // stage x_t tile (16x12 f32 -> bf16, zero-padded K=32)
    if (tid < BM_ * IN_) {
      int m = tid / IN_;
      int k = tid - m * IN_;
      xbuf[m * 32 + k] = (__bf16)xp[((size_t)(b0 + m) * T_ + t) * IN_ + k];
    }
    if (tid < BM_) {
      int tn = (t + 1 < T_) ? t + 1 : t;
      __builtin_prefetch(&xp[((size_t)(b0 + tid) * T_ + tn) * IN_], 0, 1);
    }
    __syncthreads();   // (1) x ready; h buffers from prev step stable

    v16bf aX = ldA(xrow, 0, hiH);
    v16bf aH0[4], aH1[4];
#pragma unroll
    for (int kt = 0; kt < 4; ++kt) {
      aH0[kt] = ldA(h0row, kt, hiH);
      aH1[kt] = ldA(h1row, kt, hiH);
    }

    // ---- layer 0: gates = bias + x_t@Wih0^T + h0@Whh0^T ----
    v8f acc[4];
#pragma unroll
    for (int g = 0; g < 4; ++g) {
      v8f a;
#pragma unroll
      for (int e = 0; e < 8; ++e) a[e] = bias0v[g];
      a = wmma_bf16(aX, fih0[g], a);
      const int n = g * H_ + wcol + lm;
#pragma unroll
      for (int kt = 0; kt < 4; ++kt)
        a = wmma_bf16(aH0[kt], ldB(W0, n, kt * 32 + kof), a);
      acc[g] = a;
    }
    float h0n[8];
#pragma unroll
    for (int e = 0; e < 8; ++e) {                       // i,f,g,o = acc[0..3]
      float cv = sig_(acc[1][e]) * c0[e] + sig_(acc[0][e]) * th_(acc[2][e]);
      c0[e]  = cv;
      h0n[e] = sig_(acc[3][e]) * th_(cv);
    }
    __syncthreads();   // (2) all waves done reading h0b
#pragma unroll
    for (int e = 0; e < 8; ++e)
      h0b[(rb + e) * HPAD_ + hcol] = (__bf16)h0n[e];
    __syncthreads();   // (3) h0b now holds h0_t

    // ---- layer 1: gates = bias + h0_t@Wih1^T + h1@Whh1^T ----
#pragma unroll
    for (int kt = 0; kt < 4; ++kt) aH0[kt] = ldA(h0row, kt, hiH);
#pragma unroll
    for (int g = 0; g < 4; ++g) {
      v8f a;
#pragma unroll
      for (int e = 0; e < 8; ++e) a[e] = bias1v[g];
      const int n = g * H_ + wcol + lm;
#pragma unroll
      for (int kt = 0; kt < 4; ++kt) {
        a = wmma_bf16(aH0[kt], fih1[g][kt], a);
        a = wmma_bf16(aH1[kt], ldB(W1, n, kt * 32 + kof), a);
      }
      acc[g] = a;
    }
    float h1n[8];
#pragma unroll
    for (int e = 0; e < 8; ++e) {
      float cv = sig_(acc[1][e]) * c1[e] + sig_(acc[0][e]) * th_(acc[2][e]);
      c1[e]  = cv;
      h1n[e] = sig_(acc[3][e]) * th_(cv);
    }
#pragma unroll
    for (int e = 0; e < 8; ++e)
      h1b[(rb + e) * HPAD_ + hcol] = (__bf16)h1n[e];
    // next iteration's barrier (1) orders h1b writes vs. reads
  }

  __syncthreads();
  // ---- output head: out[b,:] = h1_T @ Wout^T + bout  (tiny GEMV) ----
  if (tid < BM_ * OUT_) {
    int m = tid >> 2, o = tid & 3;
    float s = bout[o];
#pragma unroll 4
    for (int j = 0; j < H_; ++j)
      s += (float)h1b[m * HPAD_ + j] * Wout[o * H_ + j];
    out[(b0 + m) * OUT_ + o] = s;
  }
}

extern "C" void kernel_launch(void* const* d_in, const int* in_sizes, int n_in,
                              void* d_out, int out_size, void* d_ws, size_t ws_size,
                              hipStream_t stream) {
  const float* xp   = (const float*)d_in[0];
  const float* Wih0 = (const float*)d_in[1];
  const float* Whh0 = (const float*)d_in[2];
  const float* bih0 = (const float*)d_in[3];
  const float* bhh0 = (const float*)d_in[4];
  const float* Wih1 = (const float*)d_in[5];
  const float* Whh1 = (const float*)d_in[6];
  const float* bih1 = (const float*)d_in[7];
  const float* bhh1 = (const float*)d_in[8];
  const float* Wout = (const float*)d_in[9];
  const float* bout = (const float*)d_in[10];
  float* out = (float*)d_out;

  const int B = in_sizes[0] / (T_ * IN_);   // 256
  // LDS: 2 packed recurrent weight matrices + x tile + two h buffers
  const size_t shmem = (size_t)2 * (2u * G4_ * HPAD_            // W0 + W1
                                    + BM_ * 32                  // xbuf
                                    + 2u * BM_ * HPAD_);        // h0b + h1b
  (void)hipFuncSetAttribute(reinterpret_cast<const void*>(lstm2_fused),
                            hipFuncAttributeMaxDynamicSharedMemorySize,
                            (int)shmem);

  dim3 grid(B / BM_);
  dim3 block(256);
  lstm2_fused<<<grid, block, shmem, stream>>>(xp, Wih0, Whh0, bih0, bhh0,
                                              Wih1, Whh1, bih1, bhh1,
                                              Wout, bout, out);
}
